// MolTriangleUpdate_50044958933072
// MI455X (gfx1250) — matrix operand order repaired and checked by reading, not compiled, not run on hardware
//
#include <hip/hip_runtime.h>
#include <hip/hip_bf16.h>

#define NDIM 448
#define NN   (448 * 448)
#define PAIRC 128
#define HIDC  64

typedef __bf16 bf16_t;
typedef __attribute__((ext_vector_type(16))) __bf16 v16bf;
typedef __attribute__((ext_vector_type(8)))  __bf16 v8bf;
typedef __attribute__((ext_vector_type(8)))  float  v8f;
typedef __attribute__((ext_vector_type(4)))  int    v4i;

typedef __attribute__((address_space(1))) v4i as1_v4i;
typedef __attribute__((address_space(3))) v4i as3_v4i;

// Async L2->LDS copy path (CDNA5): guarded so the file compiles on either
// toolchain; fallback is plain load+ds_store.
#if defined(__gfx1250__)
#if __has_builtin(__builtin_amdgcn_global_load_async_to_lds_b128)
#define HAS_ASYNC_LDS 1
#endif
#endif

__device__ __forceinline__ void ld_tile_b128(const bf16_t* g, bf16_t* l) {
#ifdef HAS_ASYNC_LDS
  __builtin_amdgcn_global_load_async_to_lds_b128((as1_v4i*)g, (as3_v4i*)l, 0, 0);
#else
  *(v8bf*)l = *(const v8bf*)g;
#endif
}

__device__ __forceinline__ void wait_async_lds() {
#ifdef HAS_ASYNC_LDS
#if __has_builtin(__builtin_amdgcn_s_wait_asynccnt)
  __builtin_amdgcn_s_wait_asynccnt(0);
#else
  asm volatile("s_wait_asynccnt 0x0" ::: "memory");
#endif
#endif
}

// ---------------- workspace layout (bytes) ----------------
#define OFF_BLOCK  0ull
#define SZ_BLOCK   (64ull * NN * 4ull)          // block[c][i][j] f32
#define SZ_ABP     (64ull * NN * 2ull)          // one ab plane set, bf16
#define OFF_AB1N   (OFF_BLOCK + SZ_BLOCK)
#define OFF_AB1T   (OFF_AB1N + SZ_ABP)
#define OFF_AB2N   (OFF_AB1T + SZ_ABP)
#define OFF_AB2T   (OFF_AB2N + SZ_ABP)
#define OFF_GATE   (OFF_AB2T + SZ_ABP)
#define SZ_GATE    ((unsigned long long)NN * PAIRC * 2ull)
#define OFF_WCAT   (OFF_GATE + SZ_GATE)
#define SZ_WCAT    (128ull * 384ull * 2ull)
#define OFF_OUTW   (OFF_WCAT + SZ_WCAT)

// ---------------- operand loaders ----------------
// 16-bit A-matrix 16x32 layout (ISA 7.12.2): lane m (h = lane>>4):
//   halves 0..7  -> K = k0 + 8*h + 0..7
//   halves 8..15 -> K = k0 + 16 + 8*h + 0..7
__device__ __forceinline__ v16bf load_a16(const bf16_t* base, int strideH,
                                          int m, int h, int k0) {
  const v8bf lo = *(const v8bf*)(base + m * strideH + k0 + 8 * h);
  const v8bf hi = *(const v8bf*)(base + m * strideH + k0 + 16 + 8 * h);
  return __builtin_shufflevector(lo, hi, 0, 1, 2, 3, 4, 5, 6, 7,
                                 8, 9, 10, 11, 12, 13, 14, 15);
}

// B operand (32 x 16): lane n = lane%16 column; halves 0..7 -> K = 8*h + 0..7,
// halves 8..15 -> K = 16 + 8*h + 0..7 (strided gather, p at column n).
__device__ __forceinline__ v16bf load_b16(const bf16_t* p, int stride, int h) {
  v16bf r;
#pragma unroll
  for (int s = 0; s < 8; ++s) {
    r[s] = p[(8 * h + s) * stride];
    r[s + 8] = p[(16 + 8 * h + s) * stride];
  }
  return r;
}

__device__ __forceinline__ v8f wmma_bf16(v16bf a, v16bf b, v8f c) {
  return __builtin_amdgcn_wmma_f32_16x16x32_bf16(false, a, false, b,
                                                 (short)0, c, false, false);
}

__device__ __forceinline__ float sigmoidf_(float x) {
  return 1.0f / (1.0f + __expf(-x));
}

// ---------------- kernel 0: weight conversion ----------------
__global__ __launch_bounds__(256) void k_prep(
    const float* __restrict__ g1w, const float* __restrict__ l1w,
    const float* __restrict__ g2w, const float* __restrict__ l2w,
    const float* __restrict__ egw, const float* __restrict__ outw,
    bf16_t* __restrict__ wcat, bf16_t* __restrict__ outwb) {
  int t = blockIdx.x * blockDim.x + threadIdx.x;
  const int nw = 128 * 384;
  if (t < nw) {
    int k = t / 384, n = t % 384;
    float v;
    if (n < 64)       v = g1w[k * 64 + n];
    else if (n < 128) v = l1w[k * 64 + (n - 64)];
    else if (n < 192) v = g2w[k * 64 + (n - 128)];
    else if (n < 256) v = l2w[k * 64 + (n - 192)];
    else              v = egw[k * 128 + (n - 256)];
    wcat[t] = (bf16_t)v;
  } else if (t < nw + 64 * 128) {
    int u = t - nw;
    outwb[u] = (bf16_t)outw[u];
  }
}

// ---------------- kernel 1: LN + gated projections ----------------
// 64 pixels per block; 8 waves. Produces ab1N/ab2N (natural per-channel bf16
// planes, coalesced v8bf stores) and the eg sigmoid gate.
__global__ __launch_bounds__(256) void k_phase1(
    const float* __restrict__ z, const float* __restrict__ zmask,
    const float* __restrict__ lnw, const float* __restrict__ lnb,
    const float* __restrict__ g1b, const float* __restrict__ g2b,
    const float* __restrict__ l1b, const float* __restrict__ l2b,
    const float* __restrict__ egb, const bf16_t* __restrict__ wcat,
    bf16_t* __restrict__ ab1N, bf16_t* __restrict__ ab2N,
    bf16_t* __restrict__ gate) {
  __shared__ bf16_t zn[64 * 136];  // 272B stride -> conflict-free A reads
  const int tid = threadIdx.x;
  const int wave = tid >> 5, lane = tid & 31;
  const int pix0 = blockIdx.x * 64;

  // --- LayerNorm over 128 channels; each wave owns 8 pixel rows ---
  for (int r = 0; r < 8; ++r) {
    const int row = wave * 8 + r;
    const int pix = pix0 + row;
    const float* zp = z + (size_t)pix * PAIRC + lane * 4;
    float x0 = zp[0], x1 = zp[1], x2 = zp[2], x3 = zp[3];
    float s = x0 + x1 + x2 + x3;
    float q = x0 * x0 + x1 * x1 + x2 * x2 + x3 * x3;
#pragma unroll
    for (int off = 16; off >= 1; off >>= 1) {
      s += __shfl_xor(s, off, 32);
      q += __shfl_xor(q, off, 32);
    }
    const float mu = s * (1.0f / 128.0f);
    const float var = q * (1.0f / 128.0f) - mu * mu;
    const float rstd = rsqrtf(var + 1e-5f);
    const int c0 = lane * 4;
    bf16_t* dst = zn + row * 136 + c0;
    dst[0] = (bf16_t)((x0 - mu) * rstd * lnw[c0 + 0] + lnb[c0 + 0]);
    dst[1] = (bf16_t)((x1 - mu) * rstd * lnw[c0 + 1] + lnb[c0 + 1]);
    dst[2] = (bf16_t)((x2 - mu) * rstd * lnw[c0 + 2] + lnb[c0 + 2]);
    dst[3] = (bf16_t)((x3 - mu) * rstd * lnw[c0 + 3] + lnb[c0 + 3]);
  }
  __syncthreads();

  const int hi = lane >> 4, nn = lane & 15, mrow = lane & 15;

  // --- task A: paired sigmoid/linear projection (ab1: waves 0-3, ab2: 4-7)
  {
    const int isAb2 = (wave >= 4);
    const int pt = wave & 3;                       // channel tile 16*pt..
    const int gcol = (isAb2 ? 128 : 0) + pt * 16;  // sigmoid branch cols
    const int lcol = gcol + 64;                    // linear branch cols
    const float* gb = isAb2 ? g2b : g1b;
    const float* lb = isAb2 ? l2b : l1b;
    bf16_t* abN = isAb2 ? ab2N : ab1N;

    v16bf Bg[4], Bl[4];
#pragma unroll
    for (int kc = 0; kc < 4; ++kc) {
      const int k = kc * 32;
      Bg[kc] = load_b16(wcat + (size_t)k * 384 + gcol + nn, 384, hi);
      Bl[kc] = load_b16(wcat + (size_t)k * 384 + lcol + nn, 384, hi);
    }
#pragma unroll
    for (int rt = 0; rt < 4; ++rt) {
      v8f accG = {}; v8f accL = {};
#pragma unroll
      for (int kc = 0; kc < 4; ++kc) {
        const v16bf a = load_a16(zn, 136, rt * 16 + mrow, hi, kc * 32);
        accG = wmma_bf16(a, Bg[kc], accG);
        accL = wmma_bf16(a, Bl[kc], accL);
      }
      const int c = pt * 16 + nn;
      const int pixb = pix0 + rt * 16 + 8 * hi;  // lane's 8 consecutive pixels
      v8bf pk;
#pragma unroll
      for (int r = 0; r < 8; ++r) {
        const float mval = zmask[pixb + r];
        const float g = accG[r] + gb[c];
        const float l = accL[r] + lb[c];
        pk[r] = (bf16_t)(sigmoidf_(g) * l * mval);
      }
      *(v8bf*)(abN + (size_t)c * NN + pixb) = pk;  // [c][pix] 16B store
    }
  }

  // --- task B: eg gate projection (128 cols, one 16-col tile per wave) ---
  {
    const int col = 256 + wave * 16;
    v16bf Be[4];
#pragma unroll
    for (int kc = 0; kc < 4; ++kc) {
      const int k = kc * 32;
      Be[kc] = load_b16(wcat + (size_t)k * 384 + col + nn, 384, hi);
    }
#pragma unroll
    for (int rt = 0; rt < 4; ++rt) {
      v8f acc = {};
#pragma unroll
      for (int kc = 0; kc < 4; ++kc) {
        const v16bf a = load_a16(zn, 136, rt * 16 + mrow, hi, kc * 32);
        acc = wmma_bf16(a, Be[kc], acc);
      }
      const int c = wave * 16 + nn;
#pragma unroll
      for (int r = 0; r < 8; ++r) {  // coalesces across lanes (nn contiguous)
        const int pix = pix0 + rt * 16 + r + 8 * hi;
        gate[(size_t)pix * PAIRC + c] = (bf16_t)sigmoidf_(acc[r] + egb[c]);
      }
    }
  }
}

// ---------------- kernel 1b: per-channel 64x64 tiled transpose ----------
// Fills abT[c][k][i] = abN[c][i][k] with fully coalesced 16B accesses.
// grid (7, 7, 128): z<64 -> ab1, z>=64 -> ab2.
__global__ __launch_bounds__(256) void k_transpose(
    const bf16_t* __restrict__ ab1N, const bf16_t* __restrict__ ab2N,
    bf16_t* __restrict__ ab1T, bf16_t* __restrict__ ab2T) {
  __shared__ bf16_t t[64 * 72];
  const int tid = threadIdx.x;
  const int cz = blockIdx.z;
  const bf16_t* src = (cz < 64 ? ab1N : ab2N) + (size_t)(cz & 63) * NN;
  bf16_t* dst = (cz < 64 ? ab1T : ab2T) + (size_t)(cz & 63) * NN;
  const int i0 = blockIdx.y * 64, k0 = blockIdx.x * 64;

#pragma unroll
  for (int p = 0; p < 2; ++p) {
    const int row = (tid >> 3) + p * 32, seg = tid & 7;
    *(v8bf*)(t + row * 72 + seg * 8) =
        *(const v8bf*)(src + (size_t)(i0 + row) * NDIM + k0 + seg * 8);
  }
  __syncthreads();
#pragma unroll
  for (int p = 0; p < 2; ++p) {
    const int kk = (tid >> 3) + p * 32, seg = tid & 7;
    v8bf v;
#pragma unroll
    for (int u = 0; u < 8; ++u) v[u] = t[(seg * 8 + u) * 72 + kk];
    *(v8bf*)(dst + (size_t)(k0 + kk) * NDIM + i0 + seg * 8) = v;
  }
}

// ---------------- kernel 2: triangle products ----------------
// grid (7, 7, 64): 64x64 (i,j) tile of channel c.
// block[i,j,c] = ab1N*ab2T + ab1T*ab2N  (row-major A, KxN row-major B).
// Double-buffered LDS; async L2->LDS staging when available.
__global__ __launch_bounds__(256) void k_phase2(
    const bf16_t* __restrict__ ab1N, const bf16_t* __restrict__ ab1T,
    const bf16_t* __restrict__ ab2N, const bf16_t* __restrict__ ab2T,
    float* __restrict__ blockOut) {
  __shared__ bf16_t A1[2][64 * 56];  // 112B stride: conflict-free 16 rows
  __shared__ bf16_t A2[2][64 * 56];
  __shared__ bf16_t B1[2][32 * 72];  // 144B stride
  __shared__ bf16_t B2[2][32 * 72];
  const int tid = threadIdx.x;
  const int wave = tid >> 5, lane = tid & 31;
  const int c = blockIdx.z;
  const int i0 = blockIdx.y * 64, j0 = blockIdx.x * 64;
  const bf16_t* p1N = ab1N + (size_t)c * NN;
  const bf16_t* p1T = ab1T + (size_t)c * NN;
  const bf16_t* p2N = ab2N + (size_t)c * NN;
  const bf16_t* p2T = ab2T + (size_t)c * NN;

  const int rt = wave >> 1;        // row tile 0..3
  const int ct0 = (wave & 1) * 2;  // col tiles {0,1} or {2,3}
  const int hi = lane >> 4, nn = lane & 15, mrow = lane & 15;

  const int arow = tid >> 2, aseg = tid & 3;  // A tile: 64 x 32 halves
  const int brow = tid >> 3, bseg = tid & 7;  // B tile: 32 x 64 halves

  auto issue = [&](int kc, int buf) {
    const int k0 = kc * 32;
    ld_tile_b128(p1N + (size_t)(i0 + arow) * NDIM + k0 + aseg * 8,
                 &A1[buf][arow * 56 + aseg * 8]);
    ld_tile_b128(p1T + (size_t)(i0 + arow) * NDIM + k0 + aseg * 8,
                 &A2[buf][arow * 56 + aseg * 8]);
    ld_tile_b128(p2T + (size_t)(k0 + brow) * NDIM + j0 + bseg * 8,
                 &B1[buf][brow * 72 + bseg * 8]);
    ld_tile_b128(p2N + (size_t)(k0 + brow) * NDIM + j0 + bseg * 8,
                 &B2[buf][brow * 72 + bseg * 8]);
  };

  v8f acc0 = {}; v8f acc1 = {};
  issue(0, 0);

  for (int kc = 0; kc < 14; ++kc) {
    const int cur = kc & 1;
    wait_async_lds();   // my async ops landed in LDS
    __syncthreads();    // everyone's landed; prev compute done reading cur^1
    if (kc < 13) issue(kc + 1, cur ^ 1);

    const v16bf a1 = load_a16(A1[cur], 56, rt * 16 + mrow, hi, 0);
    const v16bf a2 = load_a16(A2[cur], 56, rt * 16 + mrow, hi, 0);
    const v16bf b1a = load_b16(B1[cur] + (ct0 + 0) * 16 + nn, 72, hi);
    const v16bf b2a = load_b16(B2[cur] + (ct0 + 0) * 16 + nn, 72, hi);
    const v16bf b1b = load_b16(B1[cur] + (ct0 + 1) * 16 + nn, 72, hi);
    const v16bf b2b = load_b16(B2[cur] + (ct0 + 1) * 16 + nn, 72, hi);

    acc0 = wmma_bf16(a1, b1a, acc0);
    acc0 = wmma_bf16(a2, b2a, acc0);
    acc1 = wmma_bf16(a1, b1b, acc1);
    acc1 = wmma_bf16(a2, b2b, acc1);
  }

  float* outp = blockOut + (size_t)c * NN;
#pragma unroll
  for (int r = 0; r < 8; ++r) {
    const int i = i0 + rt * 16 + r + 8 * hi;
    const int j = j0 + ct0 * 16 + nn;
    outp[(size_t)i * NDIM + j] = acc0[r];
    outp[(size_t)i * NDIM + j + 16] = acc1[r];
  }
}

// ---------------- kernel 3: LN(block) @ out_w, gate, mask ----------------
__global__ __launch_bounds__(256) void k_phase3(
    const float* __restrict__ blockIn, const bf16_t* __restrict__ gate,
    const float* __restrict__ lnhw, const float* __restrict__ lnhb,
    const bf16_t* __restrict__ outwb, const float* __restrict__ outb,
    const float* __restrict__ zmask, float* __restrict__ out) {
  __shared__ float  blk[128 * 68];
  __shared__ bf16_t hn[128 * 72];
  const int tid = threadIdx.x;
  const int wave = tid >> 5, lane = tid & 31;
  const int p0 = blockIdx.x * 128;

  // gather [64 ch][128 pix] -> LDS transposed [pix][ch]
  {
    const int cch = tid >> 2, seg = tid & 3;
    const float4* src = (const float4*)(blockIn + (size_t)cch * NN + p0 + seg * 32);
#pragma unroll
    for (int u = 0; u < 8; ++u) {
      const float4 v = src[u];
      const int pr = seg * 32 + u * 4;
      blk[(pr + 0) * 68 + cch] = v.x;
      blk[(pr + 1) * 68 + cch] = v.y;
      blk[(pr + 2) * 68 + cch] = v.z;
      blk[(pr + 3) * 68 + cch] = v.w;
    }
  }
  __syncthreads();

  if (tid < 128) {  // LN over 64 channels, one pixel per thread
    const float* row = blk + tid * 68;
    float s = 0.f, q = 0.f;
#pragma unroll
    for (int ci = 0; ci < 64; ++ci) { const float v = row[ci]; s += v; q += v * v; }
    const float mu = s * (1.0f / 64.0f);
    const float var = q * (1.0f / 64.0f) - mu * mu;
    const float rstd = rsqrtf(var + 1e-5f);
    bf16_t* hrow = hn + tid * 72;
#pragma unroll
    for (int ci = 0; ci < 64; ++ci)
      hrow[ci] = (bf16_t)((row[ci] - mu) * rstd * lnhw[ci] + lnhb[ci]);
  }
  __syncthreads();

  const int hi = lane >> 4, nn = lane & 15, mrow = lane & 15;
  const int rt = wave;  // 8 row tiles of 16 pixels
  for (int ct = 0; ct < 8; ++ct) {
    v8f acc = {};
#pragma unroll
    for (int kc = 0; kc < 2; ++kc) {
      const v16bf a = load_a16(hn, 72, rt * 16 + mrow, hi, kc * 32);
      const int k = kc * 32;
      const v16bf b = load_b16(outwb + (size_t)k * 128 + ct * 16 + nn, 128, hi);
      acc = wmma_bf16(a, b, acc);
    }
    const int oc = ct * 16 + nn;
#pragma unroll
    for (int r = 0; r < 8; ++r) {
      const int pix = p0 + rt * 16 + r + 8 * hi;
      const float g = (float)gate[(size_t)pix * PAIRC + oc];
      out[(size_t)pix * PAIRC + oc] = (acc[r] + outb[oc]) * g * zmask[pix];
    }
  }
}

// ---------------- launcher ----------------
extern "C" void kernel_launch(void* const* d_in, const int* in_sizes, int n_in,
                              void* d_out, int out_size, void* d_ws, size_t ws_size,
                              hipStream_t stream) {
  const float* z     = (const float*)d_in[0];
  const float* zmask = (const float*)d_in[1];
  const float* lnw   = (const float*)d_in[2];
  const float* lnb   = (const float*)d_in[3];
  const float* lnhw  = (const float*)d_in[4];
  const float* lnhb  = (const float*)d_in[5];
  const float* g1w   = (const float*)d_in[6];
  const float* g1b   = (const float*)d_in[7];
  const float* g2w   = (const float*)d_in[8];
  const float* g2b   = (const float*)d_in[9];
  const float* l1w   = (const float*)d_in[10];
  const float* l1b   = (const float*)d_in[11];
  const float* l2w   = (const float*)d_in[12];
  const float* l2b   = (const float*)d_in[13];
  const float* egw   = (const float*)d_in[14];
  const float* egb   = (const float*)d_in[15];
  const float* outw  = (const float*)d_in[16];
  const float* outb  = (const float*)d_in[17];

  char* ws = (char*)d_ws;
  float*  blockW = (float*)(ws + OFF_BLOCK);
  bf16_t* ab1N   = (bf16_t*)(ws + OFF_AB1N);
  bf16_t* ab1T   = (bf16_t*)(ws + OFF_AB1T);
  bf16_t* ab2N   = (bf16_t*)(ws + OFF_AB2N);
  bf16_t* ab2T   = (bf16_t*)(ws + OFF_AB2T);
  bf16_t* gateB  = (bf16_t*)(ws + OFF_GATE);
  bf16_t* wcat   = (bf16_t*)(ws + OFF_WCAT);
  bf16_t* outwb  = (bf16_t*)(ws + OFF_OUTW);

  k_prep<<<dim3((128 * 384 + 64 * 128 + 255) / 256), 256, 0, stream>>>(
      g1w, l1w, g2w, l2w, egw, outw, wcat, outwb);

  k_phase1<<<dim3(NN / 64), 256, 0, stream>>>(
      z, zmask, lnw, lnb, g1b, g2b, l1b, l2b, egb, wcat,
      ab1N, ab2N, gateB);

  k_transpose<<<dim3(7, 7, 128), 256, 0, stream>>>(ab1N, ab2N, ab1T, ab2T);

  k_phase2<<<dim3(7, 7, 64), 256, 0, stream>>>(ab1N, ab1T, ab2N, ab2T, blockW);

  k_phase3<<<dim3(NN / 128), 256, 0, stream>>>(
      blockW, gateB, lnhw, lnhb, outwb, outb, zmask, (float*)d_out);
}
